// GlobalSpatialAdaptKernelModule_19542101196951
// MI455X (gfx1250) — matrix operand order repaired and checked by reading, not compiled, not run on hardware
//
#include <hip/hip_runtime.h>
#include <hip/hip_bf16.h>

// ---------------- problem constants (from reference) ----------------
#define NB   32
#define C_IN 512
#define CM   128
#define HH   48
#define WW   48
#define HW   (HH * WW)      // 2304
#define PL   512
#define KR   9              // 3x3

typedef _Float16 half_t;
typedef __attribute__((ext_vector_type(16))) _Float16 v16h;
typedef __attribute__((ext_vector_type(8)))  _Float16 v8h;
typedef __attribute__((ext_vector_type(8)))  float    v8f;

// A-fragment loader for v_wmma_f32_16x16x32_f16.
// A 16x32 k order chosen here (not the ISA table order): lane L holds row M = L%16; half = L/16 selects
// K = half*16 + 0..15 (elements 0..15, contiguous, matching B).
__device__ __forceinline__ v16h load_a_frag(const half_t* rowptr, int c0, int halfsel) {
    union { v16h v; v8h h[2]; } u;
    u.h[0] = *(const v8h*)(rowptr + c0 + halfsel * 16);
    u.h[1] = *(const v8h*)(rowptr + c0 + halfsel * 16 + 8);
    return u.v;
}

__device__ __forceinline__ v8h zero_v8h() {
    v8h z;
#pragma unroll
    for (int i = 0; i < 8; ++i) z[i] = (_Float16)0;
    return z;
}

__device__ __forceinline__ v8f zero_v8f() {
    v8f z;
#pragma unroll
    for (int i = 0; i < 8; ++i) z[i] = 0.0f;
    return z;
}

// ---------------- P0: convert Wc / Wf to f16 ----------------
__global__ __launch_bounds__(256) void k_convert_weights(
    const float* __restrict__ Wc, const float* __restrict__ Wf,
    half_t* __restrict__ Wc16, half_t* __restrict__ Wf16) {
    int i = blockIdx.x * 256 + threadIdx.x;
    if (i < CM * C_IN) Wc16[i] = (half_t)Wc[i];
    if (i < PL * CM)   Wf16[i] = (half_t)Wf[i];
}

// ---------------- K1: global mean + 3x3 max pool per (n,c) ----------------
__global__ __launch_bounds__(256) void k_pool(
    const float* __restrict__ x, float* __restrict__ xmean, float* __restrict__ xmax) {
    int nc = blockIdx.x;                      // n*C_IN + c
    const float* p = x + (size_t)nc * HW;
    int tid = threadIdx.x;
    float s = 0.0f;
    float mx[KR];
#pragma unroll
    for (int r = 0; r < KR; ++r) mx[r] = -3.4e38f;
    for (int i = tid; i < HW; i += 256) {
        float v = p[i];
        s += v;
        int h = i / WW, w = i - h * WW;
        int r = (h >> 4) * 3 + (w >> 4);      // 16x16 pooling blocks
        mx[r] = fmaxf(mx[r], v);
    }
    __shared__ float ssum[256];
    __shared__ float smax[KR][256];
    ssum[tid] = s;
#pragma unroll
    for (int r = 0; r < KR; ++r) smax[r][tid] = mx[r];
    __syncthreads();
    for (int off = 128; off > 0; off >>= 1) {
        if (tid < off) {
            ssum[tid] += ssum[tid + off];
#pragma unroll
            for (int r = 0; r < KR; ++r)
                smax[r][tid] = fmaxf(smax[r][tid], smax[r][tid + off]);
        }
        __syncthreads();
    }
    if (tid == 0) {
        xmean[nc] = ssum[0] * (1.0f / (float)HW);
#pragma unroll
        for (int r = 0; r < KR; ++r) xmax[(size_t)nc * KR + r] = smax[r][0];
    }
}

// ---------------- K2: gk1 / g_kern / sk per (n, c-out of shared conv) ----------------
__global__ __launch_bounds__(CM) void k_branch(
    const float* __restrict__ Wc, const float* __restrict__ bc,
    const float* __restrict__ Wkk, const float* __restrict__ bkk,
    const float* __restrict__ xmean, const float* __restrict__ xmax,
    float* __restrict__ gbuf, float* __restrict__ sbuf) {
    int n = blockIdx.x;
    int c = threadIdx.x;                      // 0..127
    const float* wrow = Wc + (size_t)c * C_IN;
    const float* xm   = xmean + (size_t)n * C_IN;
    const float* xx   = xmax + (size_t)n * C_IN * KR;
    float accm = 0.0f;
    float accr[KR];
#pragma unroll
    for (int r = 0; r < KR; ++r) accr[r] = 0.0f;
    for (int ci = 0; ci < C_IN; ++ci) {
        float w = wrow[ci];
        accm += w * xm[ci];
#pragma unroll
        for (int r = 0; r < KR; ++r) accr[r] += w * xx[(size_t)ci * KR + r];
    }
    float b = bc[c];
    float gk1 = fmaxf(accm + b, 0.0f);
    size_t base = ((size_t)n * CM + c) * KR;
#pragma unroll
    for (int r = 0; r < KR; ++r) {
        sbuf[base + r] = fmaxf(accr[r] + b, 0.0f);     // sk
        gbuf[base + r] = gk1 * Wkk[r] + bkk[r];        // g_kern
    }
}

// ---------------- K3: dynamic weights dynWt[n][r][o][c] (f16) ----------------
__global__ __launch_bounds__(256) void k_dynw(
    const float* __restrict__ Wck, const float* __restrict__ bck,
    const float* __restrict__ gbuf, const float* __restrict__ sbuf,
    half_t* __restrict__ dynWt) {
    int idx = blockIdx.x * 256 + threadIdx.x;  // ((n*9 + r)*128 + o)*128 + c
    if (idx >= NB * KR * CM * CM) return;
    int c = idx & 127;
    int o = (idx >> 7) & 127;
    int r = (idx / (CM * CM)) % KR;
    int n = idx / (CM * CM * KR);
    size_t gi = ((size_t)n * CM + c) * KR + r;
    float v = Wck[o * 2 + 0] * gbuf[gi] + Wck[o * 2 + 1] * sbuf[gi] + bck[o];
    dynWt[idx] = (half_t)fmaxf(v, 0.0f);
}

// ---------------- S1: f = relu(Wc . x + bc), output channel-last f16 fT[n][p][o] ----
// One block = 16 pixels x 128 out-channels. Full K=512 strip staged in LDS once
// (16 KB), single barrier, then 16 back-to-back WMMA steps per wave.
__global__ __launch_bounds__(256) void k_fgemm(
    const float* __restrict__ x, const half_t* __restrict__ Wc16,
    const float* __restrict__ bc, half_t* __restrict__ fT) {
    __shared__ __align__(64) half_t lds[16][C_IN];   // [pixel][c] : 16 KB
    int n  = blockIdx.y;
    int p0 = blockIdx.x * 16;
    int tid = threadIdx.x;
    int wave = tid >> 5, lane = tid & 31;
    int col = lane & 15, halfsel = lane >> 4;
    int o0 = wave * 16;
    const float* xbase = x + (size_t)n * C_IN * HW + p0;

    // cooperative fill: 512 rows x 4 pixel-quads = 2048 float4 loads
#pragma unroll
    for (int e = 0; e < 8; ++e) {
        int idx = tid + e * 256;          // 0..2047
        int cc = idx >> 2;                // channel 0..511
        int q  = idx & 3;                 // pixel quad
        float4 v = *(const float4*)(xbase + (size_t)cc * HW + q * 4);
        lds[q * 4 + 0][cc] = (half_t)v.x;
        lds[q * 4 + 1][cc] = (half_t)v.y;
        lds[q * 4 + 2][cc] = (half_t)v.z;
        lds[q * 4 + 3][cc] = (half_t)v.w;
    }
    __syncthreads();

    const half_t* arow = Wc16 + (size_t)(o0 + col) * C_IN;   // A row = Wc[o0+col][:]
    v8f acc = zero_v8f();
#pragma unroll
    for (int c0 = 0; c0 < C_IN; c0 += 32) {
        v16h a = load_a_frag(arow, c0, halfsel);
        v16h b = *(const v16h*)(&lds[col][c0 + halfsel * 16]);
        acc = __builtin_amdgcn_wmma_f32_16x16x32_f16(false, a, false, b,
                                                     (short)0, acc, false, false);
    }
    // D layout: lane col = pixel, rows o = o0 + halfsel*8 + i
    union { v8h v; half_t e[8]; } out;
#pragma unroll
    for (int i = 0; i < 8; ++i) {
        int o = o0 + halfsel * 8 + i;
        out.e[i] = (half_t)fmaxf(acc[i] + bc[o], 0.0f);
    }
    *(v8h*)(fT + ((size_t)(n * HW + p0 + col)) * CM + o0 + halfsel * 8) = out.v;
}

// ---------------- S5: dynamic 3x3 conv as implicit GEMM, yT[n][p][o] f16 ----------
// Halo window of fT (3 rows x 18 pixels x 128 ch, zero-padded) staged in LDS once
// and shared by all 8 waves; inner loop is unconditional LDS vector loads + WMMA.
__global__ __launch_bounds__(256) void k_dynconv(
    const half_t* __restrict__ fT, const half_t* __restrict__ dynWt,
    const float* __restrict__ b_ad, half_t* __restrict__ yT) {
    __shared__ __align__(64) half_t bl[3][18][CM];   // 13,824 B
    int n = blockIdx.y;
    int p0 = blockIdx.x * 16;          // 16 consecutive pixels, single image row
    int h = p0 / WW;
    int w0 = p0 - h * WW;
    int tid = threadIdx.x;
    int wave = tid >> 5, lane = tid & 31;
    int col = lane & 15, halfsel = lane >> 4;
    int o0 = wave * 16;

    // cooperative halo fill: 54 (row,pixel) runs x 4 chunks of 32 halfs = 216 chunks
    if (tid < 216) {
        int j   = tid / 72;            // halo row 0..2
        int rem = tid - j * 72;
        int px  = rem >> 2;            // halo pixel 0..17
        int q   = rem & 3;             // 32-half chunk
        int hp = h + j - 1;
        int wp = w0 + px - 1;
        half_t* dst = &bl[j][px][q * 32];
        if (hp >= 0 && hp < HH && wp >= 0 && wp < WW) {
            const half_t* src = fT + ((size_t)(n * HW + hp * WW + wp)) * CM + q * 32;
            *(v8h*)(dst + 0)  = *(const v8h*)(src + 0);
            *(v8h*)(dst + 8)  = *(const v8h*)(src + 8);
            *(v8h*)(dst + 16) = *(const v8h*)(src + 16);
            *(v8h*)(dst + 24) = *(const v8h*)(src + 24);
        } else {
            v8h z = zero_v8h();
            *(v8h*)(dst + 0)  = z;
            *(v8h*)(dst + 8)  = z;
            *(v8h*)(dst + 16) = z;
            *(v8h*)(dst + 24) = z;
        }
    }
    __syncthreads();

    const half_t* dynA = dynWt + (size_t)n * KR * CM * CM;
    v8f acc = zero_v8f();
#pragma unroll
    for (int r = 0; r < KR; ++r) {
        int dh = r / 3, dw = r - dh * 3;
        int hp = h + dh - 1;
        if (hp < 0 || hp >= HH) continue;      // wave-uniform skip (zero-pad row)
        const half_t* arow = dynA + ((size_t)r * CM + (o0 + col)) * CM;
#pragma unroll
        for (int c0 = 0; c0 < CM; c0 += 32) {
            v16h a = load_a_frag(arow, c0, halfsel);
            v16h b = *(const v16h*)(&bl[dh][col + dw][c0 + halfsel * 16]);
            acc = __builtin_amdgcn_wmma_f32_16x16x32_f16(false, a, false, b,
                                                         (short)0, acc, false, false);
        }
    }
    union { v8h v; half_t e[8]; } out;
#pragma unroll
    for (int i = 0; i < 8; ++i) {
        int o = o0 + halfsel * 8 + i;
        out.e[i] = (half_t)(acc[i] + b_ad[o]);         // no relu on y
    }
    *(v8h*)(yT + ((size_t)(n * HW + p0 + col)) * CM + o0 + halfsel * 8) = out.v;
}

// ---------------- S6: out = Wf . y + bf  (f32, NCHW) ----------------------------
// yT tile (16 px x 128 ch = 4 KB) staged in LDS once, reused by all 8 waves and
// across all 4 P-groups (512 planes) handled by the same block.
__global__ __launch_bounds__(256) void k_fuse(
    const half_t* __restrict__ yT, const half_t* __restrict__ Wf16,
    const float* __restrict__ bf, float* __restrict__ out) {
    __shared__ __align__(64) half_t bl[16][CM];      // 4 KB
    int n = blockIdx.y;
    int p0 = blockIdx.x * 16;
    int tid = threadIdx.x;
    int wave = tid >> 5, lane = tid & 31;
    int col = lane & 15, halfsel = lane >> 4;

    {   // cooperative fill: 2048 halfs, one v8h per thread
        int px = tid >> 4;             // 0..15
        int q  = tid & 15;             // 16 chunks of 8 halfs
        *(v8h*)(&bl[px][q * 8]) =
            *(const v8h*)(yT + ((size_t)(n * HW + p0 + px)) * CM + q * 8);
    }
    __syncthreads();

    int p = p0 + col;
#pragma unroll
    for (int pg = 0; pg < PL / CM; ++pg) {
        int P0 = pg * CM + wave * 16;
        const half_t* arow = Wf16 + (size_t)(P0 + col) * CM;
        v8f acc = zero_v8f();
#pragma unroll
        for (int c0 = 0; c0 < CM; c0 += 32) {
            v16h a = load_a_frag(arow, c0, halfsel);
            v16h b = *(const v16h*)(&bl[col][c0 + halfsel * 16]);
            acc = __builtin_amdgcn_wmma_f32_16x16x32_f16(false, a, false, b,
                                                         (short)0, acc, false, false);
        }
#pragma unroll
        for (int i = 0; i < 8; ++i) {
            int P = P0 + halfsel * 8 + i;
            out[((size_t)n * PL + P) * HW + p] = acc[i] + bf[P];
        }
    }
}

// ---------------- host side ----------------
extern "C" void kernel_launch(void* const* d_in, const int* in_sizes, int n_in,
                              void* d_out, int out_size, void* d_ws, size_t ws_size,
                              hipStream_t stream) {
    const float* x    = (const float*)d_in[0];
    const float* Wc   = (const float*)d_in[1];
    const float* bc   = (const float*)d_in[2];
    const float* Wkk  = (const float*)d_in[3];
    const float* bkk  = (const float*)d_in[4];
    const float* Wck  = (const float*)d_in[5];
    const float* bck  = (const float*)d_in[6];
    const float* b_ad = (const float*)d_in[7];
    const float* Wf   = (const float*)d_in[8];
    const float* bf   = (const float*)d_in[9];
    float* out = (float*)d_out;

    // workspace carve-up (256B aligned)
    size_t off = 0;
    auto carve = [&](size_t bytes) -> void* {
        void* p = (char*)d_ws + off;
        off += (bytes + 255) & ~(size_t)255;
        return p;
    };
    half_t* Wc16  = (half_t*)carve((size_t)CM * C_IN * sizeof(half_t));
    half_t* Wf16  = (half_t*)carve((size_t)PL * CM * sizeof(half_t));
    float*  xmean = (float*) carve((size_t)NB * C_IN * sizeof(float));
    float*  xmax  = (float*) carve((size_t)NB * C_IN * KR * sizeof(float));
    float*  gbuf  = (float*) carve((size_t)NB * CM * KR * sizeof(float));
    float*  sbuf  = (float*) carve((size_t)NB * CM * KR * sizeof(float));
    half_t* dynWt = (half_t*)carve((size_t)NB * KR * CM * CM * sizeof(half_t));
    half_t* fT    = (half_t*)carve((size_t)NB * HW * CM * sizeof(half_t));
    half_t* yT    = (half_t*)carve((size_t)NB * HW * CM * sizeof(half_t));
    (void)ws_size; (void)n_in; (void)in_sizes; (void)out_size;

    // small prep
    k_convert_weights<<<(CM * C_IN + 255) / 256, 256, 0, stream>>>(Wc, Wf, Wc16, Wf16);
    k_pool<<<NB * C_IN, 256, 0, stream>>>(x, xmean, xmax);
    k_branch<<<NB, CM, 0, stream>>>(Wc, bc, Wkk, bkk, xmean, xmax, gbuf, sbuf);
    k_dynw<<<(NB * KR * CM * CM + 255) / 256, 256, 0, stream>>>(Wck, bck, gbuf, sbuf, dynWt);

    // big WMMA stages
    dim3 g1(HW / 16, NB);
    k_fgemm<<<g1, 256, 0, stream>>>(x, Wc16, bc, fT);
    k_dynconv<<<g1, 256, 0, stream>>>(fT, dynWt, b_ad, yT);
    k_fuse<<<g1, 256, 0, stream>>>(yT, Wf16, bf, out);
}
